// ExpertChoiceMoRLayer_28140625723544
// MI455X (gfx1250) — matrix operand order repaired and checked by reading, not compiled, not run on hardware
//
#include <hip/hip_runtime.h>

typedef unsigned short u16;
typedef unsigned int   u32;
typedef __attribute__((ext_vector_type(16))) __bf16 v16bf;
typedef __attribute__((ext_vector_type(8)))  float  v8f;
typedef __attribute__((ext_vector_type(4)))  u32    v4u;
typedef __attribute__((ext_vector_type(4)))  int    v4i;
typedef __attribute__((ext_vector_type(8)))  int    v8i;

#define B_    4
#define S_    2048
#define H_    1024
#define I_    4096
#define R_    3
#define KTOP  1024
#define NH_   16
#define DH_   64
#define ALPHA_ 0.1f

#if defined(__has_builtin)
#if __has_builtin(__builtin_amdgcn_tensor_load_to_lds) && \
    __has_builtin(__builtin_amdgcn_s_wait_tensorcnt)
#define HAVE_TDM 1
#endif
#endif
#ifndef HAVE_TDM
#define HAVE_TDM 0
#endif

union Frag { u32 u[8]; v16bf v; };

__device__ inline u16 f2bf(float f) {
  u32 u = __builtin_bit_cast(u32, f);
  u32 r = u + 0x7FFFu + ((u >> 16) & 1u);
  return (u16)(r >> 16);
}

__device__ inline v8f wmma_bf16(const Frag& a, const Frag& b, v8f c) {
  return __builtin_amdgcn_wmma_f32_16x16x32_bf16(false, a.v, false, b.v,
                                                 (short)0, c, false, false);
}

// A fragment: 16x32 bf16, row-major source with leading dim lda (elements).
// lanes 0-15 -> M=0..15; VGPR j: K=half*16+2j (contiguous order, matches B loaders)
__device__ inline void load_a(Frag& f, const u16* A, int lda, int lane) {
  int half = lane >> 4, m = lane & 15;
  const u16* base = A + m * lda;
#pragma unroll
  for (int j = 0; j < 8; ++j) {
    int kk = half * 16 + 2 * j;
    f.u[j] = *(const u32*)(base + kk);
  }
}

// B = W^T fragment (32x16): W is [N][K] row-major; B[k][n] = W[n][k].
__device__ inline void load_bt(Frag& f, const u16* W, int ldw, int lane) {
  int half = lane >> 4, n = lane & 15;
  const u16* base = W + n * ldw + half * 16;
#pragma unroll
  for (int j = 0; j < 8; ++j) f.u[j] = *(const u32*)(base + 2 * j);
}

// B fragment from row-major B[k][n] (k = rows): used for P@V (V is [key][d]).
__device__ inline void load_b_rm(Frag& f, const u16* Bm, int ldb, int lane) {
  int half = lane >> 4, n = lane & 15;
#pragma unroll
  for (int j = 0; j < 8; ++j) {
    int kk = half * 16 + 2 * j;
    u32 lo = Bm[(size_t)kk * ldb + n];
    u32 hi = Bm[(size_t)(kk + 1) * ldb + n];
    f.u[j] = lo | (hi << 16);
  }
}

#if HAVE_TDM
// TDM: DMA a 128-row x 32-element bf16 tile (row stride = rs elements) into LDS.
// D# per ISA 8.3/8.4: group0 = {count=1, lds_addr, global_addr, type=2};
// group1 = {data_size=2B, tensor_dim0=rs, tensor_dim1=128, tile=32x128, dim0_stride=rs}.
__device__ inline void tdm_load_tile_128x32(u32 lds_addr, const u16* gptr, int rs) {
  unsigned long long ga = (unsigned long long)(size_t)gptr;
  v4u g0;
  g0.x = 1u;                                   // count=1 valid descriptor
  g0.y = lds_addr;                             // LDS byte address
  g0.z = (u32)ga;                              // global_addr[31:0]
  g0.w = (u32)((ga >> 32) & 0x01FFFFFFu) | (2u << 30);  // addr[56:32] | type=2
  u32 urs = (u32)rs;
  v8i g1 = {0, 0, 0, 0, 0, 0, 0, 0};
  g1[0] = (int)(1u << 16);                     // data_size = 1 (2 bytes)
  g1[1] = (int)((urs & 0xFFFFu) << 16);        // tensor_dim0[15:0]
  g1[2] = (int)(((urs >> 16) & 0xFFFFu) | (128u << 16)); // dim0 hi | tensor_dim1 lo
  g1[3] = (int)(32u << 16);                    // tensor_dim1 hi=0 | tile_dim0=32
  g1[4] = (int)128u;                           // tile_dim1=128, tile_dim2=0
  g1[5] = (int)urs;                            // tensor_dim0_stride[31:0]
  v4i z4 = {0, 0, 0, 0};
#if __clang_major__ >= 23
  v8i z8 = {0, 0, 0, 0, 0, 0, 0, 0};
  __builtin_amdgcn_tensor_load_to_lds(g0, g1, z4, z4, z8, 0);
#else
  __builtin_amdgcn_tensor_load_to_lds(g0, g1, z4, z4, 0);
#endif
}
#endif

// Load 128x32 bf16 A tile into LDS (TDM if available, else per-thread loads).
__device__ inline void stage_a_tile(u32* AsU, const u16* A, size_t row0, int Kd,
                                    int kofs, int tid, int wave) {
#if HAVE_TDM
  if (wave == 0) {
    tdm_load_tile_128x32((u32)(size_t)AsU, A + row0 * Kd + kofs, Kd);
    __builtin_amdgcn_s_wait_tensorcnt((short)0);
  }
#else
  for (int e = tid; e < 2048; e += 256) {
    int row = e >> 4, cp = e & 15;
    AsU[e] = *(const u32*)(A + (row0 + row) * Kd + kofs + 2 * cp);
  }
#endif
}

// ---------------------------------------------------------------- copy
__global__ __launch_bounds__(256) void copy_kernel(const float4* __restrict__ src,
                                                   float4* __restrict__ dst, int n) {
  int i = blockIdx.x * 256 + threadIdx.x;
  if (i < n) dst[i] = src[i];
}

// ---------------------------------------------------------------- router
__global__ __launch_bounds__(256) void router_kernel(const float* __restrict__ hs,
                                                     const float* __restrict__ Wr,
                                                     float* __restrict__ logits,
                                                     float* __restrict__ probs) {
  int gw = (blockIdx.x << 3) + (threadIdx.x >> 5);
  int lane = threadIdx.x & 31;
  if (gw >= R_ * B_ * S_) return;
  int r = gw / (B_ * S_);
  int bs = gw - r * (B_ * S_);
  const float* x = hs + (size_t)bs * H_;
  const float* w = Wr + (size_t)r * H_;
  float acc = 0.f;
  for (int i = lane; i < H_; i += 32) acc += x[i] * w[i];
  for (int m = 16; m; m >>= 1) acc += __shfl_xor(acc, m);
  if (lane == 0) {
    logits[gw] = acc;
    probs[gw] = ALPHA_ * __builtin_amdgcn_rcpf(1.f + __expf(-acc));
  }
}

// ---------------------------------------------------------------- top-k (rank based, matches jax tie-break)
__global__ __launch_bounds__(256) void topk_kernel(const float* __restrict__ probs,
                                                   int* __restrict__ idx,
                                                   float* __restrict__ wsel, int r) {
  __shared__ float p[S_];
  __shared__ int rk[S_];
  int b = blockIdx.x;
  const float* pr = probs + ((size_t)r * B_ + b) * S_;
  for (int i = threadIdx.x; i < S_; i += 256) p[i] = pr[i];
  __syncthreads();
  for (int i = threadIdx.x; i < S_; i += 256) {
    float pi = p[i];
    int c = 0;
    for (int j = 0; j < S_; ++j) {
      float pj = p[j];
      c += (pj > pi) || (pj == pi && j < i);
    }
    rk[i] = c;
  }
  __syncthreads();
  for (int i = threadIdx.x; i < S_; i += 256) {
    if (rk[i] < KTOP) {
      int pos = 0;
      for (int j = 0; j < i; ++j) pos += (rk[j] < KTOP);
      idx[(b << 10) + pos] = i;
      wsel[(b << 10) + pos] = p[i];
    }
  }
}

// ---------------------------------------------------------------- gather + LN1
__global__ __launch_bounds__(256) void gather_ln_kernel(const float* __restrict__ hs,
                                                        const int* __restrict__ idx,
                                                        const float* __restrict__ lw,
                                                        const float* __restrict__ lb,
                                                        float* __restrict__ Xsel,
                                                        u16* __restrict__ Aln) {
  int gw = (blockIdx.x << 3) + (threadIdx.x >> 5);
  int lane = threadIdx.x & 31;
  if (gw >= B_ * KTOP) return;
  int b = gw >> 10, kk = gw & 1023;
  int tok = idx[(b << 10) + kk];
  const float* x = hs + ((size_t)b * S_ + tok) * H_;
  float xr[32];
  float s = 0.f, s2 = 0.f;
#pragma unroll
  for (int i = 0; i < 32; ++i) {
    float t = x[(i << 5) + lane];
    xr[i] = t; s += t; s2 += t * t;
  }
  for (int m = 16; m; m >>= 1) { s += __shfl_xor(s, m); s2 += __shfl_xor(s2, m); }
  float mu = s * (1.f / H_);
  float var = s2 * (1.f / H_) - mu * mu;
  float rstd = rsqrtf(var + 1e-6f);
  float* xo = Xsel + (size_t)gw * H_;
  u16* ao = Aln + (size_t)gw * H_;
#pragma unroll
  for (int i = 0; i < 32; ++i) {
    int hh = (i << 5) + lane;
    float t = xr[i];
    xo[hh] = t;
    ao[hh] = f2bf((t - mu) * rstd * lw[hh] + lb[hh]);
  }
}

// ---------------------------------------------------------------- LN2
__global__ __launch_bounds__(256) void ln2_kernel(const float* __restrict__ X,
                                                  const float* __restrict__ lw,
                                                  const float* __restrict__ lb,
                                                  u16* __restrict__ Aln) {
  int gw = (blockIdx.x << 3) + (threadIdx.x >> 5);
  int lane = threadIdx.x & 31;
  if (gw >= B_ * KTOP) return;
  const float* x = X + (size_t)gw * H_;
  float xr[32];
  float s = 0.f, s2 = 0.f;
#pragma unroll
  for (int i = 0; i < 32; ++i) {
    float t = x[(i << 5) + lane];
    xr[i] = t; s += t; s2 += t * t;
  }
  for (int m = 16; m; m >>= 1) { s += __shfl_xor(s, m); s2 += __shfl_xor(s2, m); }
  float mu = s * (1.f / H_);
  float var = s2 * (1.f / H_) - mu * mu;
  float rstd = rsqrtf(var + 1e-6f);
  u16* ao = Aln + (size_t)gw * H_;
#pragma unroll
  for (int i = 0; i < 32; ++i) {
    int hh = (i << 5) + lane;
    ao[hh] = f2bf((xr[i] - mu) * rstd * lw[hh] + lb[hh]);
  }
}

// ---------------------------------------------------------------- generic WMMA GEMM: C[M,N] = A_bf16[M,Kd] @ W_f32[N,Kd]^T
// EP 0: store bf16 to outb ; EP 1: resid[m,n] += acc ; EP 2: total scatter-add
template <int EP>
__global__ __launch_bounds__(256) void gemm_kernel(const u16* __restrict__ A,
                                                   const float* __restrict__ W,
                                                   int M, int N, int Kd,
                                                   u16* __restrict__ outb,
                                                   float* __restrict__ resid,
                                                   float* __restrict__ total,
                                                   const int* __restrict__ idx,
                                                   const float* __restrict__ wsel) {
  __shared__ u32 AsU[128 * 16];  // 128 x 32 bf16
  __shared__ u32 WsU[64 * 16];   // 64  x 32 bf16
  int tid = threadIdx.x, lane = tid & 31, wave = tid >> 5;
  int wm = wave >> 1, wn = wave & 1;
  int m0 = blockIdx.x * 128, n0 = blockIdx.y * 64;
  v8f acc[2][2] = {};
  int nk = Kd >> 5;
  for (int kt = 0; kt < nk; ++kt) {
    stage_a_tile(AsU, A, (size_t)m0, Kd, kt * 32, tid, wave);
    for (int e = tid; e < 1024; e += 256) {
      int row = e >> 4, cp = e & 15;
      const float* wp = W + (size_t)(n0 + row) * Kd + kt * 32 + 2 * cp;
      WsU[row * 16 + cp] = (u32)f2bf(wp[0]) | ((u32)f2bf(wp[1]) << 16);
    }
    if (kt + 1 < nk) __builtin_prefetch(W + (size_t)n0 * Kd + (kt + 1) * 32, 0, 1);
    __syncthreads();
    const u16* As = (const u16*)AsU;
    const u16* Ws = (const u16*)WsU;
    Frag a0, a1, b0, b1;
    load_a(a0, As + (wm * 32 + 0) * 32, 32, lane);
    load_a(a1, As + (wm * 32 + 16) * 32, 32, lane);
    load_bt(b0, Ws + (wn * 32 + 0) * 32, 32, lane);
    load_bt(b1, Ws + (wn * 32 + 16) * 32, 32, lane);
    acc[0][0] = wmma_bf16(a0, b0, acc[0][0]);
    acc[0][1] = wmma_bf16(a0, b1, acc[0][1]);
    acc[1][0] = wmma_bf16(a1, b0, acc[1][0]);
    acc[1][1] = wmma_bf16(a1, b1, acc[1][1]);
    __syncthreads();
  }
  int half = lane >> 4, nn = lane & 15;
#pragma unroll
  for (int i = 0; i < 2; ++i)
#pragma unroll
    for (int j = 0; j < 2; ++j) {
      int mt = m0 + wm * 32 + i * 16;
      int nt = n0 + wn * 32 + j * 16;
#pragma unroll
      for (int vv = 0; vv < 8; ++vv) {
        int m = mt + vv + 8 * half;
        int n = nt + nn;
        float val = acc[i][j][vv];
        if (EP == 0) {
          outb[(size_t)m * N + n] = f2bf(val);
        } else if (EP == 1) {
          resid[(size_t)m * N + n] += val;
        } else {
          int b = m >> 10, kk = m & 1023;
          int tok = idx[(b << 10) + kk];
          float w = wsel[(b << 10) + kk];
          total[((size_t)b * S_ + tok) * H_ + n] +=
              (resid[(size_t)m * H_ + n] + val) * w;
        }
      }
    }
}

// ---------------------------------------------------------------- fused gate/up GEMM + SiLU: act = silu(A@Wg^T) * (A@Wu^T)
__global__ __launch_bounds__(256) void gateup_kernel(const u16* __restrict__ A,
                                                     const float* __restrict__ Wg,
                                                     const float* __restrict__ Wu,
                                                     u16* __restrict__ act,
                                                     int M, int N, int Kd) {
  __shared__ u32 AsU[128 * 16];
  __shared__ u32 GsU[64 * 16];
  __shared__ u32 UsU[64 * 16];
  int tid = threadIdx.x, lane = tid & 31, wave = tid >> 5;
  int wm = wave >> 1, wn = wave & 1;
  int m0 = blockIdx.x * 128, n0 = blockIdx.y * 64;
  v8f ag[2][2] = {}, au[2][2] = {};
  int nk = Kd >> 5;
  for (int kt = 0; kt < nk; ++kt) {
    stage_a_tile(AsU, A, (size_t)m0, Kd, kt * 32, tid, wave);
    for (int e = tid; e < 1024; e += 256) {
      int row = e >> 4, cp = e & 15;
      size_t off = (size_t)(n0 + row) * Kd + kt * 32 + 2 * cp;
      const float* gp = Wg + off;
      const float* up = Wu + off;
      GsU[row * 16 + cp] = (u32)f2bf(gp[0]) | ((u32)f2bf(gp[1]) << 16);
      UsU[row * 16 + cp] = (u32)f2bf(up[0]) | ((u32)f2bf(up[1]) << 16);
    }
    __syncthreads();
    const u16* As = (const u16*)AsU;
    const u16* Gs = (const u16*)GsU;
    const u16* Us = (const u16*)UsU;
    Frag a0, a1, bg0, bg1, bu0, bu1;
    load_a(a0, As + (wm * 32 + 0) * 32, 32, lane);
    load_a(a1, As + (wm * 32 + 16) * 32, 32, lane);
    load_bt(bg0, Gs + (wn * 32 + 0) * 32, 32, lane);
    load_bt(bg1, Gs + (wn * 32 + 16) * 32, 32, lane);
    load_bt(bu0, Us + (wn * 32 + 0) * 32, 32, lane);
    load_bt(bu1, Us + (wn * 32 + 16) * 32, 32, lane);
    ag[0][0] = wmma_bf16(a0, bg0, ag[0][0]);
    ag[0][1] = wmma_bf16(a0, bg1, ag[0][1]);
    ag[1][0] = wmma_bf16(a1, bg0, ag[1][0]);
    ag[1][1] = wmma_bf16(a1, bg1, ag[1][1]);
    au[0][0] = wmma_bf16(a0, bu0, au[0][0]);
    au[0][1] = wmma_bf16(a0, bu1, au[0][1]);
    au[1][0] = wmma_bf16(a1, bu0, au[1][0]);
    au[1][1] = wmma_bf16(a1, bu1, au[1][1]);
    __syncthreads();
  }
  int half = lane >> 4, nn = lane & 15;
#pragma unroll
  for (int i = 0; i < 2; ++i)
#pragma unroll
    for (int j = 0; j < 2; ++j) {
      int mt = m0 + wm * 32 + i * 16;
      int nt = n0 + wn * 32 + j * 16;
#pragma unroll
      for (int vv = 0; vv < 8; ++vv) {
        int m = mt + vv + 8 * half;
        int n = nt + nn;
        float g = ag[i][j][vv];
        float u = au[i][j][vv];
        float silu = g * __builtin_amdgcn_rcpf(1.f + __expf(-g));
        act[(size_t)m * N + n] = f2bf(silu * u);
      }
    }
}

// ---------------------------------------------------------------- causal flash attention (wave = 16 query rows)
__global__ __launch_bounds__(128) void attn_kernel(const u16* __restrict__ Q,
                                                   const u16* __restrict__ Kt,
                                                   const u16* __restrict__ V,
                                                   u16* __restrict__ O) {
  __shared__ u32 Pu[4][256];  // per-wave 16x32 bf16 prob tile
  int lane = threadIdx.x & 31, wave = threadIdx.x >> 5;
  int bid = blockIdx.x;
  int qb = bid & 15;
  int h = (bid >> 4) & 15;
  int b = bid >> 8;
  int q0 = qb * 64 + wave * 16;
  size_t rowbase = (size_t)b * KTOP;
  int col = h * DH_;
  int half = lane >> 4, nn = lane & 15;

  Frag aq[2];
  load_a(aq[0], Q + (rowbase + q0) * H_ + col, H_, lane);
  load_a(aq[1], Q + (rowbase + q0) * H_ + col + 32, H_, lane);

  int nkt = (q0 >> 4) + 1;
  float rm[8], rs[8];
#pragma unroll
  for (int v = 0; v < 8; ++v) { rm[v] = -3.0e38f; rs[v] = 0.f; }

  // pass 1: row max / sum(exp)
  for (int kt = 0; kt < nkt; ++kt) {
    v8f s = {};
    Frag bk;
    load_bt(bk, Kt + (rowbase + kt * 16) * H_ + col, H_, lane);
    s = wmma_bf16(aq[0], bk, s);
    load_bt(bk, Kt + (rowbase + kt * 16) * H_ + col + 32, H_, lane);
    s = wmma_bf16(aq[1], bk, s);
#pragma unroll
    for (int v = 0; v < 8; ++v) {
      int qg = q0 + v + 8 * half;
      int kg = kt * 16 + nn;
      float sv = (kg <= qg) ? s[v] * 0.125f : -3.0e38f;
      float tm = sv;
      for (int m = 8; m; m >>= 1) tm = fmaxf(tm, __shfl_xor(tm, m));
      float nm = fmaxf(rm[v], tm);
      float e = __expf(sv - nm);
      float es = e;
      for (int m = 8; m; m >>= 1) es += __shfl_xor(es, m);
      rs[v] = rs[v] * __expf(rm[v] - nm) + es;
      rm[v] = nm;
    }
  }
  // reciprocal of row sums, hoisted out of the P-tile loop
#pragma unroll
  for (int v = 0; v < 8; ++v) rs[v] = __builtin_amdgcn_rcpf(rs[v]);

  // pass 2: P @ V
  v8f oacc[4] = {};
  int npair = (nkt + 1) >> 1;
  for (int kp = 0; kp < npair; ++kp) {
    u16* P = (u16*)Pu[wave];
#pragma unroll
    for (int sub = 0; sub < 2; ++sub) {
      int kt = kp * 2 + sub;
      if (kt < nkt) {
        v8f s = {};
        Frag bk;
        load_bt(bk, Kt + (rowbase + kt * 16) * H_ + col, H_, lane);
        s = wmma_bf16(aq[0], bk, s);
        load_bt(bk, Kt + (rowbase + kt * 16) * H_ + col + 32, H_, lane);
        s = wmma_bf16(aq[1], bk, s);
#pragma unroll
        for (int v = 0; v < 8; ++v) {
          int qg = q0 + v + 8 * half;
          int kg = kt * 16 + nn;
          float sv = (kg <= qg) ? s[v] * 0.125f : -3.0e38f;
          float p = __expf(sv - rm[v]) * rs[v];
          P[(v + 8 * half) * 32 + sub * 16 + nn] = f2bf(p);
        }
      } else {
#pragma unroll
        for (int v = 0; v < 8; ++v)
          P[(v + 8 * half) * 32 + sub * 16 + nn] = 0;
      }
    }
    asm volatile("s_wait_dscnt 0" ::: "memory");
    Frag ap;
    load_a(ap, (const u16*)Pu[wave], 32, lane);
#pragma unroll
    for (int dt = 0; dt < 4; ++dt) {
      Frag bv;
      load_b_rm(bv, V + (rowbase + kp * 32) * H_ + col + dt * 16, H_, lane);
      oacc[dt] = wmma_bf16(ap, bv, oacc[dt]);
    }
  }

#pragma unroll
  for (int dt = 0; dt < 4; ++dt)
#pragma unroll
    for (int v = 0; v < 8; ++v) {
      int m = q0 + v + 8 * half;
      O[(rowbase + m) * H_ + col + dt * 16 + nn] = f2bf(oacc[dt][v]);
    }
}

// ---------------------------------------------------------------- launcher
extern "C" void kernel_launch(void* const* d_in, const int* in_sizes, int n_in,
                              void* d_out, int out_size, void* d_ws, size_t ws_size,
                              hipStream_t stream) {
  const float* hs   = (const float*)d_in[0];
  const float* Wr   = (const float*)d_in[1];
  const float* Wq   = (const float*)d_in[2];
  const float* Wk   = (const float*)d_in[3];
  const float* Wv   = (const float*)d_in[4];
  const float* Wo   = (const float*)d_in[5];
  const float* Wg   = (const float*)d_in[6];
  const float* Wu   = (const float*)d_in[7];
  const float* Wd   = (const float*)d_in[8];
  const float* ln1w = (const float*)d_in[9];
  const float* ln1b = (const float*)d_in[10];
  const float* ln2w = (const float*)d_in[11];
  const float* ln2b = (const float*)d_in[12];

  float* total  = (float*)d_out;
  float* logits = total + (size_t)B_ * S_ * H_;

  char* ws = (char*)d_ws;
  size_t off = 0;
  auto alloc = [&](size_t bytes) -> void* {
    void* p = ws + off;
    off += (bytes + 255) & ~(size_t)255;
    return p;
  };
  float* probs = (float*)alloc((size_t)R_ * B_ * S_ * 4);
  int*   idx   = (int*)alloc((size_t)B_ * KTOP * 4);
  float* wsel  = (float*)alloc((size_t)B_ * KTOP * 4);
  float* Xsel  = (float*)alloc((size_t)B_ * KTOP * H_ * 4);
  u16*   Aln   = (u16*)alloc((size_t)B_ * KTOP * H_ * 2);
  u16*   qb    = (u16*)alloc((size_t)B_ * KTOP * H_ * 2);
  u16*   kb    = (u16*)alloc((size_t)B_ * KTOP * H_ * 2);
  u16*   vb    = (u16*)alloc((size_t)B_ * KTOP * H_ * 2);
  u16*   ab    = (u16*)alloc((size_t)B_ * KTOP * H_ * 2);
  u16*   act   = (u16*)alloc((size_t)B_ * KTOP * I_ * 2);

  int M = B_ * KTOP;  // 4096 compacted tokens

  copy_kernel<<<(B_ * S_ * H_ / 4 + 255) / 256, 256, 0, stream>>>(
      (const float4*)hs, (float4*)total, B_ * S_ * H_ / 4);
  router_kernel<<<R_ * B_ * S_ / 8, 256, 0, stream>>>(hs, Wr, logits, probs);

  for (int r = 0; r < R_; ++r) {
    topk_kernel<<<B_, 256, 0, stream>>>(probs, idx, wsel, r);
    gather_ln_kernel<<<M / 8, 256, 0, stream>>>(hs, idx, ln1w + r * H_,
                                                ln1b + r * H_, Xsel, Aln);
    dim3 gH(M / 128, H_ / 64);
    gemm_kernel<0><<<gH, 256, 0, stream>>>(Aln, Wq + (size_t)r * H_ * H_, M, H_,
                                           H_, qb, nullptr, nullptr, nullptr,
                                           nullptr);
    gemm_kernel<0><<<gH, 256, 0, stream>>>(Aln, Wk + (size_t)r * H_ * H_, M, H_,
                                           H_, kb, nullptr, nullptr, nullptr,
                                           nullptr);
    gemm_kernel<0><<<gH, 256, 0, stream>>>(Aln, Wv + (size_t)r * H_ * H_, M, H_,
                                           H_, vb, nullptr, nullptr, nullptr,
                                           nullptr);
    attn_kernel<<<B_ * NH_ * (KTOP / 64), 128, 0, stream>>>(qb, kb, vb, ab);
    gemm_kernel<1><<<gH, 256, 0, stream>>>(ab, Wo + (size_t)r * H_ * H_, M, H_,
                                           H_, nullptr, Xsel, nullptr, nullptr,
                                           nullptr);
    ln2_kernel<<<M / 8, 256, 0, stream>>>(Xsel, ln2w + r * H_, ln2b + r * H_, Aln);
    dim3 gI(M / 128, I_ / 64);
    gateup_kernel<<<gI, 256, 0, stream>>>(Aln, Wg + (size_t)r * I_ * H_,
                                          Wu + (size_t)r * I_ * H_, act, M, I_, H_);
    gemm_kernel<2><<<gH, 256, 0, stream>>>(act, Wd + (size_t)r * H_ * I_, M, H_,
                                           I_, nullptr, Xsel, total, idx, wsel);
  }
  (void)in_sizes; (void)n_in; (void)out_size; (void)ws_size;
}
